// SimpleNewGRU_67284957659688
// MI455X (gfx1250) — matrix operand order repaired and checked by reading, not compiled, not run on hardware
//
#include <hip/hip_runtime.h>

// Problem sizes (fixed by the reference)
#define B_ 256
#define T_ 128
#define D_ 256
#define H_ 1024
#define O_ 256

#define ZROW 1288   // padded LDS row stride (1280 + 8): row bases stagger 4 banks
#define HROW 1032   // padded LDS row stride (1024 + 8)
#define NBLK 128    // persistent grid size

typedef __attribute__((ext_vector_type(16))) __bf16  v16bf;
typedef __attribute__((ext_vector_type(8)))  float   v8f;
typedef __attribute__((ext_vector_type(4)))  unsigned int u32x4;

union FragBF {
    v16bf bf;
    u32x4 q[2];
};

__device__ __forceinline__ unsigned short f2bf(float f) {
    unsigned u = __builtin_bit_cast(unsigned, f);
    u += 0x7FFFu + ((u >> 16) & 1u);          // round-to-nearest-even
    return (unsigned short)(u >> 16);
}

// A-fragment (16x32 bf16, M x K): lane L holds row M = L%16.
// lanes 0-15:  elems 0-7 -> K0+0..7,   elems 8-15 -> K0+16..23
// lanes 16-31: elems 0-7 -> K0+8..15,  elems 8-15 -> K0+24..31
__device__ __forceinline__ v16bf load_a(const unsigned short* p0) {
    FragBF f;
    f.q[0] = *(const u32x4*)(p0);
    f.q[1] = *(const u32x4*)(p0 + 16);
    return f.bf;
}

// B-fragment from pre-packed weights: 512 bf16 per 32x16 tile, lane-major.
__device__ __forceinline__ v16bf load_b(const unsigned short* p) {
    FragBF f;
    const u32x4* q = (const u32x4*)p;
    f.q[0] = q[0];
    f.q[1] = q[1];
    return f.bf;
}

// Device-wide split barrier: one arrival counter per step (pre-zeroed each call).
__device__ __forceinline__ void grid_barrier(unsigned int* cnt) {
    __syncthreads();
    if (threadIdx.x == 0) {
        __threadfence();   // release: make this block's h_new visible
        __hip_atomic_fetch_add(cnt, 1u, __ATOMIC_RELEASE, __HIP_MEMORY_SCOPE_AGENT);
        while (__hip_atomic_load(cnt, __ATOMIC_ACQUIRE, __HIP_MEMORY_SCOPE_AGENT) < (unsigned)NBLK) {
            __builtin_amdgcn_s_sleep(2);
        }
        __threadfence();   // acquire
    }
    __syncthreads();
}

// ---------------- setup kernels ----------------

__global__ void k_cvt(const float* __restrict__ src, unsigned short* __restrict__ dst, int n) {
    int i = blockIdx.x * blockDim.x + threadIdx.x;
    if (i < n) dst[i] = f2bf(src[i]);
}

// Pack row-major W[Krows][Ncols] (fp32) into WMMA B-tile layout (bf16).
__global__ void k_pack(const float* __restrict__ W, unsigned short* __restrict__ P,
                       int Krows, int Ncols) {
    int i = blockIdx.x * blockDim.x + threadIdx.x;
    int total = Krows * Ncols;
    if (i >= total) return;
    int tile = i >> 9;
    int r    = i & 511;
    int lane = r >> 4;
    int e    = r & 15;
    int ntiles = Ncols >> 4;
    int kt = tile / ntiles;
    int nt = tile - kt * ntiles;
    int K = kt * 32 + ((lane >> 4) << 3) + (e & 7) + ((e >> 3) << 4);
    int N = nt * 16 + (lane & 15);
    P[i] = f2bf(W[(size_t)K * Ncols + N]);
}

__global__ void k_inith(const float* __restrict__ h0, float* __restrict__ hF,
                        unsigned short* __restrict__ hB, int n) {
    int i = blockIdx.x * blockDim.x + threadIdx.x;
    if (i < n) {
        float v = h0[i];
        hF[i] = v;
        hB[i] = f2bf(v);
    }
}

__global__ void k_zero(unsigned int* __restrict__ p, int n) {
    int i = blockIdx.x * blockDim.x + threadIdx.x;
    if (i < n) p[i] = 0u;
}

// ---------------- persistent recurrent kernel ----------------
// 128 resident blocks, one device-wide barrier per timestep.
// Phase A (all 128 blocks): gates GEMM + sigmoid blend -> h_new (ping-pong).
//   block -> (mt = blk>>3, ng-group = (blk&7)*4 + wave); z-slab staged in LDS.
// Phase B (blocks 0..63): y = h_new @ Wo + bo -> out[:, t, :].
// Ordering: A(t) -> barrier(t) -> B(t); A(t+1) only conflicts with A(t)/B(t)
// through buffers already separated by barrier(t), so one barrier/step suffices.
__global__ __launch_bounds__(128) void k_gru(
    const unsigned short* __restrict__ xbf,   // [B][T][D] bf16
    const unsigned short* __restrict__ wcp,   // packed [1280/32][1024/16] tiles
    const unsigned short* __restrict__ wfp,
    const unsigned short* __restrict__ wop,   // packed [1024/32][256/16] tiles
    const float* __restrict__ bc,
    const float* __restrict__ bfv,
    const float* __restrict__ bo,
    float* __restrict__ hF,                   // 2 ping-pong [B][H] f32
    unsigned short* __restrict__ hB,          // 2 ping-pong [B][H] bf16
    float* __restrict__ out,                  // [B][T][O]
    unsigned int* __restrict__ bar)           // T_ arrival counters (zeroed)
{
    __shared__ __align__(16) unsigned short shz[16 * ZROW];

    const int tid  = threadIdx.x;
    const int lane = tid & 31;
    const int wave = tid >> 5;
    const int blk  = blockIdx.x;
    const int mt   = blk >> 3;                    // 0..15 batch tile (phase A)
    const int ng   = (blk & 7) * 4 + wave;        // 0..31 hidden group (32 wide)
    const int mrow = lane & 15;
    const int kb   = (lane >> 4) << 3;            // 0 or 8
    const size_t BH = (size_t)B_ * H_;

    const size_t nt0  = (size_t)(ng * 2);
    const size_t nt1  = nt0 + 1;
    const size_t lofs = (size_t)lane * 16;

    // phase-B mapping (blocks 0..63): 4 waves share one batch tile
    const int mto = blk >> 2;                     // 0..15
    const int nto = (blk & 3) * 4 + wave;         // 0..15

    for (int t = 0; t < T_; ++t) {
        const float*          hprevF = hF + (size_t)(t & 1) * BH;
        const unsigned short* hprevB = hB + (size_t)(t & 1) * BH;
        float*                hnextF = hF + (size_t)((t + 1) & 1) * BH;
        unsigned short*       hnextB = hB + (size_t)((t + 1) & 1) * BH;

        // ---------------- Phase A: gates ----------------
        __syncthreads();   // previous phase-B readers of shz are done
        // stage z = [x_t, h] slab: 16 rows x 1280; 2560 uint4 / 128 thr = 20 iters
        for (int i = tid; i < 16 * ((D_ + H_) / 8); i += 128) {
            const int r = i / 160;
            const int c = i - r * 160;
            const int k = c * 8;
            const int row = mt * 16 + r;
            const unsigned short* src = (k < D_)
                ? xbf    + (size_t)row * (T_ * D_) + (size_t)t * D_ + k
                : hprevB + (size_t)row * H_ + (k - D_);
            *(u32x4*)(&shz[r * ZROW + k]) = *(const u32x4*)src;
        }
        __syncthreads();

        v8f accC0 = {}, accC1 = {}, accF0 = {}, accF1 = {};
        const unsigned short* arow = shz + (size_t)mrow * ZROW + kb;

        #define WTB(Wp, kc, nt) ((Wp) + ((size_t)(kc) * (H_ / 16) + (nt)) * 512 + lofs)
        v16bf cC0 = load_b(WTB(wcp, 0, nt0));
        v16bf cC1 = load_b(WTB(wcp, 0, nt1));
        v16bf cF0 = load_b(WTB(wfp, 0, nt0));
        v16bf cF1 = load_b(WTB(wfp, 0, nt1));

        #pragma unroll 4
        for (int kc = 0; kc < 40; ++kc) {
            const int kn = (kc + 1 < 40) ? kc + 1 : kc;
            v16bf nC0 = load_b(WTB(wcp, kn, nt0));
            v16bf nC1 = load_b(WTB(wcp, kn, nt1));
            v16bf nF0 = load_b(WTB(wfp, kn, nt0));
            v16bf nF1 = load_b(WTB(wfp, kn, nt1));
            v16bf a = load_a(arow + kc * 32);
            accC0 = __builtin_amdgcn_wmma_f32_16x16x32_bf16(false, a, false, cC0, (short)0, accC0, false, false);
            accC1 = __builtin_amdgcn_wmma_f32_16x16x32_bf16(false, a, false, cC1, (short)0, accC1, false, false);
            accF0 = __builtin_amdgcn_wmma_f32_16x16x32_bf16(false, a, false, cF0, (short)0, accF0, false, false);
            accF1 = __builtin_amdgcn_wmma_f32_16x16x32_bf16(false, a, false, cF1, (short)0, accF1, false, false);
            cC0 = nC0; cC1 = nC1; cF0 = nF0; cF1 = nF1;
        }
        #undef WTB

        // epilogue: bias, sigmoid, blend, write h_new (f32 + bf16)
        {
            const int col0  = ng * 32 + (lane & 15);
            const int mbase = mt * 16 + ((lane >> 4) << 3);
            #pragma unroll
            for (int v = 0; v < 8; ++v) {
                const int m = mbase + v;
                {
                    const int col = col0;
                    float c  = accC0[v] + bc[col];
                    float fp = accF0[v] + bfv[col];
                    float fg = 1.0f / (1.0f + __expf(-fp));
                    float hv = hprevF[(size_t)m * H_ + col];
                    float hn = (c - hv) * fg + hv;
                    hnextF[(size_t)m * H_ + col] = hn;
                    hnextB[(size_t)m * H_ + col] = f2bf(hn);
                }
                {
                    const int col = col0 + 16;
                    float c  = accC1[v] + bc[col];
                    float fp = accF1[v] + bfv[col];
                    float fg = 1.0f / (1.0f + __expf(-fp));
                    float hv = hprevF[(size_t)m * H_ + col];
                    float hn = (c - hv) * fg + hv;
                    hnextF[(size_t)m * H_ + col] = hn;
                    hnextB[(size_t)m * H_ + col] = f2bf(hn);
                }
            }
        }

        // ---------------- device-wide barrier for step t ----------------
        grid_barrier(bar + t);

        // ---------------- Phase B: y = h_new @ Wo + bo (blocks 0..63) ----------------
        if (blk < 64) {
            // stage 16 x 1024 h_new slab (reuse shz with HROW stride)
            for (int i = tid; i < 16 * (H_ / 8); i += 128) {
                const int r = i >> 7;
                const int c = i & 127;
                *(u32x4*)(&shz[r * HROW + c * 8]) =
                    *(const u32x4*)(hnextB + (size_t)(mto * 16 + r) * H_ + c * 8);
            }
            __syncthreads();

            v8f acc = {};
            const unsigned short* orow = shz + (size_t)mrow * HROW + kb;
            v16bf cb = load_b(wop + (size_t)nto * 512 + lofs);
            #pragma unroll 4
            for (int kc = 0; kc < 32; ++kc) {
                const int kn = (kc + 1 < 32) ? kc + 1 : kc;
                v16bf nb = load_b(wop + ((size_t)kn * (O_ / 16) + nto) * 512 + lofs);
                v16bf a  = load_a(orow + kc * 32);
                acc = __builtin_amdgcn_wmma_f32_16x16x32_bf16(false, a, false, cb, (short)0, acc, false, false);
                cb = nb;
            }

            const int col   = nto * 16 + (lane & 15);
            const int mbase = mto * 16 + ((lane >> 4) << 3);
            #pragma unroll
            for (int v = 0; v < 8; ++v) {
                const int b = mbase + v;
                out[((size_t)b * T_ + t) * O_ + col] = acc[v] + bo[col];
            }
        }
    }
}

// ---------------- host side ----------------

extern "C" void kernel_launch(void* const* d_in, const int* in_sizes, int n_in,
                              void* d_out, int out_size, void* d_ws, size_t ws_size,
                              hipStream_t stream) {
    const float* x  = (const float*)d_in[0];  // [B,T,D]
    const float* h0 = (const float*)d_in[1];  // [B,H]
    const float* Wc = (const float*)d_in[2];  // [D+H,H]
    const float* bc = (const float*)d_in[3];  // [H]
    const float* Wf = (const float*)d_in[4];  // [D+H,H]
    const float* bf = (const float*)d_in[5];  // [H]
    const float* Wo = (const float*)d_in[6];  // [H,O]
    const float* bo = (const float*)d_in[7];  // [O]
    float* out = (float*)d_out;

    // Workspace layout (~24.5 MB total)
    char* w = (char*)d_ws;
    unsigned short* xbf = (unsigned short*)w;                   // B*T*D
    unsigned short* wcp = xbf + (size_t)B_ * T_ * D_;           // (D+H)*H
    unsigned short* wfp = wcp + (size_t)(D_ + H_) * H_;         // (D+H)*H
    unsigned short* wop = wfp + (size_t)(D_ + H_) * H_;         // H*O
    float*          hF  = (float*)(wop + (size_t)H_ * O_);      // 2 * B*H f32
    unsigned short* hBuf = (unsigned short*)(hF + 2 * (size_t)B_ * H_); // 2 * B*H bf16
    unsigned int*   bar = (unsigned int*)(hBuf + 2 * (size_t)B_ * H_);  // T_ counters

    int n;
    n = B_ * T_ * D_;
    k_cvt<<<(n + 255) / 256, 256, 0, stream>>>(x, xbf, n);
    n = (D_ + H_) * H_;
    k_pack<<<(n + 255) / 256, 256, 0, stream>>>(Wc, wcp, D_ + H_, H_);
    k_pack<<<(n + 255) / 256, 256, 0, stream>>>(Wf, wfp, D_ + H_, H_);
    n = H_ * O_;
    k_pack<<<(n + 255) / 256, 256, 0, stream>>>(Wo, wop, H_, O_);
    n = B_ * H_;
    k_inith<<<(n + 255) / 256, 256, 0, stream>>>(h0, hF, hBuf, n);
    k_zero<<<1, T_, 0, stream>>>(bar, T_);

    // one persistent launch runs the whole recurrence
    k_gru<<<NBLK, 128, 0, stream>>>(xbf, wcp, wfp, wop, bc, bf, bo,
                                    hF, hBuf, out, bar);
}
